// GotenNet_13176959664207
// MI455X (gfx1250) — matrix operand order repaired and checked by reading, not compiled, not run on hardware
//
#include <hip/hip_runtime.h>
#include <hip/hip_bf16.h>

// GotenNet fused attention layer for MI455X (gfx1250, wave32, WMMA).
// All heavy GEMMs run on v_wmma_f32_16x16x32_bf16 with f32 accumulation.
// Tile staging into LDS uses gfx1250 async DMA (global_load_async_to_lds_b128,
// ASYNCcnt) instead of the VGPR round-trip; next-tile HBM stream is prefetched.

typedef __attribute__((ext_vector_type(16))) __bf16 v16bf;
typedef __attribute__((ext_vector_type(8)))  float  v8f;

#define DEVINL static __device__ __forceinline__

// ---------------- scalar helpers ----------------
DEVINL unsigned short f2bf(float x) {
    unsigned u = __float_as_uint(x);
    return (unsigned short)((u + 0x7FFFu + ((u >> 16) & 1u)) >> 16);  // RNE
}
DEVINL float bf2f(unsigned short s) { return __uint_as_float(((unsigned)s) << 16); }
DEVINL float silu_f(float x) { return x / (1.f + __expf(-x)); }

// gfx1250 async copy: one 16-byte chunk, global -> LDS, no VGPR data path.
// ldsAddr is the low-32 LDS byte address (flat-address low bits), gAddr global.
DEVINL void async_g2l_b128(unsigned ldsAddr, unsigned long long gAddr) {
    asm volatile("global_load_async_to_lds_b128 %0, %1, off"
                 :: "v"(ldsAddr), "v"(gAddr) : "memory");
}
DEVINL void wait_asynccnt0() {
    asm volatile("s_wait_asynccnt 0x0" ::: "memory");
}
DEVINL unsigned lds_addr_of(const void* p) {
    return (unsigned)(unsigned long long)p;  // low 32 bits = LDS offset
}

// ---------------- WMMA fragment loads ----------------
// A 16x32 bf16 (MxK): lane l holds row M=l&15; elements [0..7]=K in [8h,8h+8),
// elements [8..15]=K in [16+8h,16+8h+8), h=l>>4.  (ISA 7.12.2)
DEVINL v16bf load_a_frag(const unsigned short* rowbase, int half) {
    union { v16bf v; uint4 q[2]; } f;
    f.q[0] = *(const uint4*)(rowbase + 8 * half);
    f.q[1] = *(const uint4*)(rowbase + 16 + 8 * half);
    return f.v;
}
// B 32x16 bf16 (KxN): lane l holds column N=l&15; elements [0..7]=K in [8h,8h+8),
// elements [8..15]=K in [16+8h,16+8h+8).  Bt is stored [N][K] row-major, so a B column = a row of Bt.
DEVINL v16bf load_b_frag(const unsigned short* colbase, int half) {
    union { v16bf v; uint4 q[2]; } f;
    const unsigned short* p = colbase + 8 * half;
    f.q[0] = *(const uint4*)(p);
    f.q[1] = *(const uint4*)(p + 16);
    return f.v;
}

// ---------------- generic 16-row-tile bf16 GEMM ----------------
// D = act( A(16*rowTiles x K) @ Bt^T + bias ),  Bt: [Nout][K] bf16 row-major.
// Output tiles are distributed round-robin over the block's waves; tile counts
// in all call sites divide evenly by 8 waves (uniform EXEC for WMMA).
DEVINL void gemm_bf16(const unsigned short* A, int lda, int rowTiles,
                      const unsigned short* Bt, int K, int Nout,
                      const float* bias, int act,
                      float* outF, int ostrF,
                      unsigned short* outB, int ostrB)
{
    const int lane = threadIdx.x & 31;
    const int wv   = threadIdx.x >> 5;
    const int nw   = blockDim.x >> 5;
    const int half = lane >> 4, l15 = lane & 15;
    const int nT = Nout >> 4, total = rowTiles * nT;
    for (int tile = wv; tile < total; tile += nw) {
        const int rt = tile / nT, nt = tile % nT;
        v8f acc = {0.f, 0.f, 0.f, 0.f, 0.f, 0.f, 0.f, 0.f};
        const unsigned short* arow = A + (size_t)(rt * 16 + l15) * lda;
        const unsigned short* brow = Bt + (size_t)(nt * 16 + l15) * K;
        for (int k0 = 0; k0 < K; k0 += 32) {
            v16bf a = load_a_frag(arow + k0, half);
            v16bf b = load_b_frag(brow + k0, half);
            acc = __builtin_amdgcn_wmma_f32_16x16x32_bf16(
                false, a, false, b, (short)0, acc, false, false);
        }
        const int col   = nt * 16 + l15;
        const float bv  = bias ? bias[col] : 0.f;
        const int rbase = rt * 16 + 8 * half;  // D: VGPR r -> row r + 8*(l>>4)
#pragma unroll
        for (int r = 0; r < 8; ++r) {
            float v = acc[r] + bv;
            if (act) v = silu_f(v);
            const int row = rbase + r;
            if (outF) outF[(size_t)row * ostrF + col] = v;
            if (outB) outB[(size_t)row * ostrB + col] = f2bf(v);
        }
    }
}

// ---------------- problem constants ----------------
constexpr int BB = 2, NN = 128, DIMC = 128, HH = 8, DHC = 32;
constexpr int DIC = 256, DMC = 512, SC = 5, SDIC = 1280;
constexpr int BN = BB * NN;

// ---------------- kernel 0a: f32 -> bf16 copy ----------------
__global__ __launch_bounds__(256) void k_conv_bf16(const float* __restrict__ s,
                                                   unsigned short* __restrict__ d, int n) {
    for (int i = blockIdx.x * blockDim.x + threadIdx.x; i < n; i += gridDim.x * blockDim.x)
        d[i] = f2bf(s[i]);
}

// ---------------- kernel 0b: f32 [K][N] -> bf16 [N][K] transpose ----------------
__global__ __launch_bounds__(256) void k_tconv(const float* __restrict__ s,
                                               unsigned short* __restrict__ d, int K, int Nn) {
    const int tot = K * Nn;
    for (int i = blockIdx.x * blockDim.x + threadIdx.x; i < tot; i += gridDim.x * blockDim.x) {
        const int k = i / Nn, n = i % Nn;
        d[(size_t)n * K + k] = f2bf(s[i]);
    }
}

// ---------------- kernel 1: node precompute ----------------
// Per 16-row tile of (b,n): LayerNorm -> hi/hj bf16; q,k,v,pv (WMMA); gates.
__global__ __launch_bounds__(256) void k_node(
    const float* __restrict__ h, const float* __restrict__ g_hi, const float* __restrict__ g_hj,
    const unsigned short* Wq_t, const unsigned short* Wk_t,
    const unsigned short* Wv1_t, const float* bv1, const unsigned short* Wv2_t, const float* bv2,
    const unsigned short* Wpv1_t, const float* bpv1, const unsigned short* Wpv2_t, const float* bpv2,
    const float* __restrict__ Wg, const float* __restrict__ bg,
    float* q_ws, float* k_ws, float* v_ws, float* pv_ws, float* gates_ws)
{
    __shared__ float hrow[16 * 128];
    __shared__ float mu[16], rinv[16];
    __shared__ unsigned short hi_bf[16 * 128];
    __shared__ unsigned short hj_bf[16 * 128];
    __shared__ unsigned short tmp_bf[16 * 512];

    const int t = threadIdx.x;
    const int r0 = blockIdx.x * 16;

    for (int i = t; i < 2048; i += 256) hrow[i] = h[(size_t)r0 * 128 + i];
    __syncthreads();
    if (t < 16) {
        float m = 0.f;
        for (int d = 0; d < 128; ++d) m += hrow[t * 128 + d];
        m *= (1.f / 128.f);
        float v = 0.f;
        for (int d = 0; d < 128; ++d) { float x = hrow[t * 128 + d] - m; v += x * x; }
        v *= (1.f / 128.f);
        mu[t] = m; rinv[t] = rsqrtf(v + 1e-5f);
    }
    __syncthreads();
    for (int i = t; i < 2048; i += 256) {
        const int r = i >> 7, d = i & 127;
        const float xn = (hrow[i] - mu[r]) * rinv[r];
        hi_bf[i] = f2bf(xn * g_hi[d]);
        hj_bf[i] = f2bf(xn * g_hj[d]);
    }
    __syncthreads();

    gemm_bf16(hi_bf, 128, 1, Wq_t, 128, 256, nullptr, 0, q_ws + (size_t)r0 * 256, 256, nullptr, 0);
    gemm_bf16(hj_bf, 128, 1, Wk_t, 128, 256, nullptr, 0, k_ws + (size_t)r0 * 256, 256, nullptr, 0);
    __syncthreads();
    gemm_bf16(hj_bf, 128, 1, Wv1_t, 128, 512, bv1, 1, nullptr, 0, tmp_bf, 512);
    __syncthreads();
    gemm_bf16(tmp_bf, 512, 1, Wv2_t, 512, 1280, bv2, 0, v_ws + (size_t)r0 * 1280, 1280, nullptr, 0);
    __syncthreads();
    gemm_bf16(hj_bf, 128, 1, Wpv1_t, 128, 512, bpv1, 1, nullptr, 0, tmp_bf, 512);
    __syncthreads();
    gemm_bf16(tmp_bf, 512, 1, Wpv2_t, 512, 1280, bpv2, 0, pv_ws + (size_t)r0 * 1280, 1280, nullptr, 0);
    __syncthreads();

    // gates = sigmoid(hi @ Wg + bg)  (128->40, VALU; tiny)
    for (int u = t; u < 16 * 40; u += 256) {
        const int r = u / 40, c = u % 40;
        float s = bg[c];
        for (int d = 0; d < 128; ++d) s += bf2f(hi_bf[r * 128 + d]) * Wg[d * 40 + c];
        gates_ws[(size_t)(r0 + r) * 40 + c] = 1.f / (1.f + __expf(-s));
    }
}

// ---------------- kernel 2: fused edge pipeline ----------------
// One workgroup per (b,i). Pass 1: ek GEMM -> sim; softmax over j in LDS.
// Pass 2: ev GEMM (recompute) -> sea -> Wo GEMM -> h/x1/x2 reductions.
constexpr int EDGE_SMEM = 16 * 1280 * 4   // big (ek|ev|out)
                        + 80 * 256 * 2    // sea bf16
                        + 16 * 256 * 4    // k_tile
                        + 128 * 40 * 4    // sim/attn
                        + 16 * 128 * 2    // t_tile bf16
                        + 256 * 4         // q_row
                        + 1280 * 4        // pv_row
                        + 64 * 4          // g_row (padded)
                        + 128 * 4         // h_acc
                        + 384 * 4         // x1_acc
                        + 640 * 4;        // x2_acc

__global__ __launch_bounds__(256) void k_edge(
    const float* __restrict__ hsrc, const unsigned short* __restrict__ t_bf,
    const float* __restrict__ r1, const float* __restrict__ r2,
    const float* __restrict__ x1, const float* __restrict__ x2,
    const unsigned short* Wek_t, const unsigned short* Wev_t, const unsigned short* Wo_t,
    const float* __restrict__ q_ws, const float* __restrict__ k_ws,
    const float* __restrict__ v_ws, const float* __restrict__ pv_ws,
    const float* __restrict__ gates_ws,
    float* __restrict__ h_out, float* __restrict__ x1_out, float* __restrict__ x2_out)
{
    extern __shared__ char smem[];
    char* p = smem;
    float*          big    = (float*)p;           p += 16 * 1280 * 4;
    unsigned short* sea    = (unsigned short*)p;  p += 80 * 256 * 2;
    float*          k_tile = (float*)p;           p += 16 * 256 * 4;
    float*          sim    = (float*)p;           p += 128 * 40 * 4;
    unsigned short* t_tile = (unsigned short*)p;  p += 16 * 128 * 2;
    float*          q_row  = (float*)p;           p += 256 * 4;
    float*          pv_row = (float*)p;           p += 1280 * 4;
    float*          g_row  = (float*)p;           p += 64 * 4;
    float*          h_acc  = (float*)p;           p += 128 * 4;
    float*          x1_acc = (float*)p;           p += 384 * 4;
    float*          x2_acc = (float*)p;           p += 640 * 4;

    const int t = threadIdx.x;
    const int b = blockIdx.x >> 7, i = blockIdx.x & 127;
    const size_t bi = (size_t)(b * 128 + i);

    for (int c = t; c < 256;  c += 256) q_row[c]  = q_ws[bi * 256 + c];
    for (int c = t; c < 1280; c += 256) pv_row[c] = pv_ws[bi * 1280 + c];
    if (t < 40)  g_row[t] = gates_ws[bi * 40 + t];
    if (t < 128) h_acc[t] = 0.f;
    for (int c = t; c < 384; c += 256) x1_acc[c] = 0.f;
    for (int c = t; c < 640; c += 256) x2_acc[c] = 0.f;
    __syncthreads();

    const unsigned short* trow = t_bf + bi * 128 * 128;  // t_ij[b,i,:,:] in bf16
    const unsigned ldsT = lds_addr_of(t_tile);
    const unsigned ldsK = lds_addr_of(k_tile);

    // ---- pass 1: ek = silu(t @ Wek), sim[j,h,s] = sum_dh q*k*ek ----
    for (int jt = 0; jt < 8; ++jt) {
        // async DMA staging: t_tile = 4KB (1 x b128/lane), k_tile = 16KB (4 x b128/lane)
        async_g2l_b128(ldsT + (unsigned)t * 16u,
                       (unsigned long long)((const char*)trow + (size_t)jt * 4096 + t * 16));
        {
            const char* ksrc = (const char*)(k_ws + (size_t)(b * 128 + jt * 16) * 256);
#pragma unroll
            for (int c = 0; c < 4; ++c)
                async_g2l_b128(ldsK + (unsigned)t * 64u + c * 16u,
                               (unsigned long long)(ksrc + t * 64 + c * 16));
        }
        // prefetch next j-tile of the only HBM-resident stream while we compute
        if (jt < 7)
            __builtin_prefetch((const char*)trow + (size_t)(jt + 1) * 4096 + t * 16, 0, 1);
        wait_asynccnt0();
        __syncthreads();

        gemm_bf16(t_tile, 128, 1, Wek_t, 128, 1280, nullptr, /*silu*/1, big, 1280, nullptr, 0);
        __syncthreads();
        for (int u = t; u < 640; u += 256) {  // (jl,h,s) units
            const int jl = u / 40, hs = u % 40, hh = hs / 5, s = hs % 5;
            float acc = 0.f;
            const int cb = hh * 32;
            for (int dh = 0; dh < 32; ++dh) {
                const int c = cb + dh;
                acc += q_row[c] * k_tile[jl * 256 + c] * big[jl * 1280 + s * 256 + c];
            }
            sim[(jt * 16 + jl) * 40 + hs] = acc;
        }
        __syncthreads();
    }

    // ---- softmax over j per (h,s) ----
    if (t < 40) {
        float m = -3.4e38f;
        for (int j = 0; j < 128; ++j) m = fmaxf(m, sim[j * 40 + t]);
        float ssum = 0.f;
        for (int j = 0; j < 128; ++j) { float e = __expf(sim[j * 40 + t] - m); sim[j * 40 + t] = e; ssum += e; }
        const float r = 1.f / ssum;
        for (int j = 0; j < 128; ++j) sim[j * 40 + t] *= r;
    }
    __syncthreads();

    // ---- pass 2: ev = t @ Wev ; sea ; out = sea @ Wo ; reductions ----
    for (int jt = 0; jt < 8; ++jt) {
        async_g2l_b128(ldsT + (unsigned)t * 16u,
                       (unsigned long long)((const char*)trow + (size_t)jt * 4096 + t * 16));
        if (jt < 7)
            __builtin_prefetch((const char*)trow + (size_t)(jt + 1) * 4096 + t * 16, 0, 1);
        wait_asynccnt0();
        __syncthreads();

        gemm_bf16(t_tile, 128, 1, Wev_t, 128, 1280, nullptr, 0, big, 1280, nullptr, 0);
        __syncthreads();
        // sea[(jl,s), h*32+dh] = (attn*v + ev*pv) * gate   -> bf16
        for (int u = t; u < 80 * 256; u += 256) {
            const int row = u >> 8, c = u & 255;
            const int jl = row / 5, s = row % 5, hh = c >> 5;
            const int j = jt * 16 + jl;
            const float a  = sim[j * 40 + hh * 5 + s];
            const float vv = v_ws[(size_t)(b * 128 + j) * 1280 + s * 256 + c];
            const float ev = big[jl * 1280 + s * 256 + c];
            sea[row * 256 + c] = f2bf((a * vv + ev * pv_row[s * 256 + c]) * g_row[hh * 5 + s]);
        }
        __syncthreads();
        gemm_bf16(sea, 256, 5, Wo_t, 256, 128, nullptr, 0, big, 128, nullptr, 0);  // 80x128 out
        __syncthreads();
        if (t < 128) {
            const int d = t;
            float ha = 0.f, a1[3] = {0, 0, 0}, a2[5] = {0, 0, 0, 0, 0};
            for (int jl = 0; jl < 16; ++jl) {
                const int j = jt * 16 + jl;
                const float* orow = big + (jl * 5) * 128;  // rows (jl,s)
                const float o0 = orow[d],        o1 = orow[128 + d], o2 = orow[256 + d];
                const float o3 = orow[384 + d],  o4 = orow[512 + d];
                ha += o0;
                const size_t xb1 = ((size_t)(b * 128 + j) * 128 + d) * 3;
                const size_t rb1 = ((size_t)(b * 128 + i) * 128 + j) * 3;
                for (int m = 0; m < 3; ++m) a1[m] += x1[xb1 + m] * o3 + r1[rb1 + m] * o1;
                const size_t xb2 = ((size_t)(b * 128 + j) * 128 + d) * 5;
                const size_t rb2 = ((size_t)(b * 128 + i) * 128 + j) * 5;
                for (int m = 0; m < 5; ++m) a2[m] += x2[xb2 + m] * o4 + r2[rb2 + m] * o2;
            }
            h_acc[d] += ha;
            for (int m = 0; m < 3; ++m) x1_acc[d * 3 + m] += a1[m];
            for (int m = 0; m < 5; ++m) x2_acc[d * 5 + m] += a2[m];
        }
        __syncthreads();
    }

    if (t < 128) {
        const int d = t;
        h_out[bi * 128 + d] = hsrc[bi * 128 + d] + h_acc[d];
        for (int m = 0; m < 3; ++m) x1_out[(bi * 128 + d) * 3 + m] = x1_acc[d * 3 + m];
        for (int m = 0; m < 5; ++m) x2_out[(bi * 128 + d) * 5 + m] = x2_acc[d * 5 + m];
    }
}

// ---------------- host launcher ----------------
extern "C" void kernel_launch(void* const* d_in, const int* in_sizes, int n_in,
                              void* d_out, int out_size, void* d_ws, size_t ws_size,
                              hipStream_t stream) {
    const float* h    = (const float*)d_in[0];
    const float* tij  = (const float*)d_in[1];
    const float* r1   = (const float*)d_in[2];
    const float* r2   = (const float*)d_in[3];
    const float* x1   = (const float*)d_in[4];
    const float* x2   = (const float*)d_in[5];
    const float* g_hi = (const float*)d_in[6];
    const float* g_hj = (const float*)d_in[7];
    const float* Wq   = (const float*)d_in[8];
    const float* Wk   = (const float*)d_in[9];
    const float* Wv1  = (const float*)d_in[10];
    const float* bv1  = (const float*)d_in[11];
    const float* Wv2  = (const float*)d_in[12];
    const float* bv2  = (const float*)d_in[13];
    const float* Wpv1 = (const float*)d_in[14];
    const float* bpv1 = (const float*)d_in[15];
    const float* Wpv2 = (const float*)d_in[16];
    const float* bpv2 = (const float*)d_in[17];
    const float* Wek  = (const float*)d_in[18];
    const float* Wev  = (const float*)d_in[19];
    const float* Wg   = (const float*)d_in[20];
    const float* bg   = (const float*)d_in[21];
    const float* Wo   = (const float*)d_in[22];

    float* out    = (float*)d_out;
    float* h_out  = out;                       // B*N*DIM          = 32768
    float* x1_out = out + 32768;               // B*N*DIM*M1       = 98304
    float* x2_out = out + 32768 + 98304;       // B*N*DIM*M2       = 163840

    char* ws = (char*)d_ws;
    size_t off = 0;
    auto carve = [&](size_t bytes) -> char* {
        char* q = ws + off;
        off = (off + bytes + 255) & ~(size_t)255;
        return q;
    };
    unsigned short* t_bf   = (unsigned short*)carve((size_t)BN * NN * DIMC * 2);  // 8.4 MB
    unsigned short* Wq_t   = (unsigned short*)carve((size_t)256 * 128 * 2);
    unsigned short* Wk_t   = (unsigned short*)carve((size_t)256 * 128 * 2);
    unsigned short* Wv1_t  = (unsigned short*)carve((size_t)512 * 128 * 2);
    unsigned short* Wv2_t  = (unsigned short*)carve((size_t)1280 * 512 * 2);
    unsigned short* Wpv1_t = (unsigned short*)carve((size_t)512 * 128 * 2);
    unsigned short* Wpv2_t = (unsigned short*)carve((size_t)1280 * 512 * 2);
    unsigned short* Wek_t  = (unsigned short*)carve((size_t)1280 * 128 * 2);
    unsigned short* Wev_t  = (unsigned short*)carve((size_t)1280 * 128 * 2);
    unsigned short* Wo_t   = (unsigned short*)carve((size_t)128 * 256 * 2);
    float* q_ws     = (float*)carve((size_t)BN * 256 * 4);
    float* k_ws     = (float*)carve((size_t)BN * 256 * 4);
    float* v_ws     = (float*)carve((size_t)BN * 1280 * 4);
    float* pv_ws    = (float*)carve((size_t)BN * 1280 * 4);
    float* gates_ws = (float*)carve((size_t)BN * 40 * 4);
    (void)ws_size; (void)n_in; (void)in_sizes; (void)out_size;

    // kernel 0: precision/layout prep (all ws is L2-resident afterwards)
    k_conv_bf16<<<2048, 256, 0, stream>>>(tij, t_bf, BN * NN * DIMC);
    auto T = [&](const float* s, unsigned short* d, int K, int Nn) {
        const int tot = K * Nn;
        k_tconv<<<(tot + 255) / 256, 256, 0, stream>>>(s, d, K, Nn);
    };
    T(Wq,   Wq_t,   128, 256);
    T(Wk,   Wk_t,   128, 256);
    T(Wv1,  Wv1_t,  128, 512);
    T(Wv2,  Wv2_t,  512, 1280);
    T(Wpv1, Wpv1_t, 128, 512);
    T(Wpv2, Wpv2_t, 512, 1280);
    T(Wek,  Wek_t,  128, 1280);
    T(Wev,  Wev_t,  128, 1280);
    T(Wo,   Wo_t,   256, 128);

    // kernel 1: node-level precompute (16 tiles of 16 rows)
    k_node<<<16, 256, 0, stream>>>(h, g_hi, g_hj, Wq_t, Wk_t,
                                   Wv1_t, bv1, Wv2_t, bv2, Wpv1_t, bpv1, Wpv2_t, bpv2,
                                   Wg, bg, q_ws, k_ws, v_ws, pv_ws, gates_ws);

    // kernel 2: fused per-(b,i) edge pipeline (needs >64KB dynamic LDS)
    (void)hipFuncSetAttribute((const void*)k_edge,
                              hipFuncAttributeMaxDynamicSharedMemorySize, EDGE_SMEM);
    k_edge<<<BN, 256, EDGE_SMEM, stream>>>(h, t_bf, r1, r2, x1, x2,
                                           Wek_t, Wev_t, Wo_t,
                                           q_ws, k_ws, v_ws, pv_ws, gates_ws,
                                           h_out, x1_out, x2_out);
}
